// ItemEmbeddingLayer_1717986918538
// MI455X (gfx1250) — matrix operand order repaired and checked by reading, not compiled, not run on hardware
//
#include <hip/hip_runtime.h>
#include <cstdint>
#include <cstddef>

#define HID 256
#define OUTD 128
#define MT 64   // rows per workgroup

typedef _Float16 h8   __attribute__((ext_vector_type(8)));
typedef _Float16 h16  __attribute__((ext_vector_type(16)));
typedef float    f8   __attribute__((ext_vector_type(8)));
typedef uint32_t u32x4 __attribute__((ext_vector_type(4)));
typedef int32_t  i32x4 __attribute__((ext_vector_type(4)));
typedef int32_t  i32x8 __attribute__((ext_vector_type(8)));

// ---------------------------------------------------------------- utilities

__global__ void zero_f32_kernel(float* __restrict__ p, size_t n) {
  size_t i = (size_t)blockIdx.x * blockDim.x + threadIdx.x;
  size_t stride = (size_t)gridDim.x * blockDim.x;
  for (; i < n; i += stride) p[i] = 0.0f;
}

__global__ void zero_last_row_kernel(float* __restrict__ out, int n) {
  out[(size_t)(n - 1) * OUTD + threadIdx.x] = 0.0f;
}

// W (f32, [din][dout]) -> WT (f16, [dout][din]) : B-fragment friendly layout
__global__ void transpose_w_kernel(const float* __restrict__ src,
                                   _Float16* __restrict__ dst,
                                   int din, int dout) {
  int idx = blockIdx.x * 256 + threadIdx.x;
  if (idx >= din * dout) return;
  int r = idx / dout, c = idx - r * dout;
  dst[(size_t)c * din + r] = (_Float16)src[idx];
}

// segment_sum(feat[src[e]]) into agg[seg[e]] with hardware f32 atomics
template <int DIM>
__global__ void scatter_add_kernel(const float* __restrict__ feat,
                                   const int* __restrict__ seg,
                                   const int* __restrict__ src,
                                   float* __restrict__ agg, int E) {
  constexpr int G = DIM / 4;
  int t = blockIdx.x * 256 + threadIdx.x;
  int e = t / G;
  if (e >= E) return;
  int g = t - e * G;
  int d = seg[e];
  int s = src[e];
  const float4 v = ((const float4*)(feat + (size_t)s * DIM))[g];
  float* a = agg + (size_t)d * DIM + (size_t)g * 4;
  __hip_atomic_fetch_add(a + 0, v.x, __ATOMIC_RELAXED, __HIP_MEMORY_SCOPE_AGENT);
  __hip_atomic_fetch_add(a + 1, v.y, __ATOMIC_RELAXED, __HIP_MEMORY_SCOPE_AGENT);
  __hip_atomic_fetch_add(a + 2, v.z, __ATOMIC_RELAXED, __HIP_MEMORY_SCOPE_AGENT);
  __hip_atomic_fetch_add(a + 3, v.w, __ATOMIC_RELAXED, __HIP_MEMORY_SCOPE_AGENT);
}

// ---------------------------------------------------------------- TDM weight stage
//
// DMA one weight K-chunk (DO rows x 64 halves, row stride DI halves) from
// global WT[DO][DI] into LDS [DO][64] with a single tensor_load_to_lds.
// Descriptor per CDNA5 ISA ch.8: data_size=4B, tile 32x DO, tensor dim0/stride
// = DI/2, type=2 ("image"). Tracked on TENSORcnt.
template <int DI, int DO>
__device__ inline void tdm_load_chunk(const _Float16* __restrict__ wtG,
                                      _Float16* __restrict__ ldsDst, int kc) {
  uint64_t gaddr = (uint64_t)(uintptr_t)(wtG + (size_t)kc * 64);
  uint32_t laddr = (uint32_t)(uintptr_t)ldsDst;   // low 32 bits = LDS byte offset
  constexpr uint32_t dim0 = DI / 2;               // row length in 4-byte units

  u32x4 g0;
  g0[0] = 1u;                                               // count=1 (valid), user mode
  g0[1] = laddr;                                            // lds_addr
  g0[2] = (uint32_t)gaddr;                                  // global_addr[31:0]
  g0[3] = ((uint32_t)(gaddr >> 32) & 0x01FFFFFFu) | (2u << 30); // addr[56:32] | type=2

  i32x8 g1;
  g1[0] = (int32_t)(2u << 16);                              // wg_mask=0, data_size=4B
  g1[1] = (int32_t)((dim0 & 0xFFFFu) << 16);                // tensor_dim0[15:0]
  g1[2] = (int32_t)((dim0 >> 16) | ((uint32_t)(DO & 0xFFFF) << 16)); // dim0 hi | tensor_dim1 lo
  g1[3] = (int32_t)(32u << 16);                             // tensor_dim1 hi=0 | tile_dim0=32
  g1[4] = (int32_t)(DO & 0xFFFF);                           // tile_dim1=DO, tile_dim2=0
  g1[5] = (int32_t)dim0;                                    // tensor_dim0_stride[31:0]
  g1[6] = 0;                                                // stride hi | dim1_stride lo
  g1[7] = 0;

  i32x4 z4 = {0, 0, 0, 0};
#if defined(__clang_major__) && (__clang_major__ >= 23)
  i32x8 z8 = {0, 0, 0, 0, 0, 0, 0, 0};
  __builtin_amdgcn_tensor_load_to_lds(g0, g1, z4, z4, z8, 0);
#else
  __builtin_amdgcn_tensor_load_to_lds(g0, g1, z4, z4, 0);
#endif
}

// ---------------------------------------------------------------- WMMA GEMM core
//
// Block = 256 threads = 8 waves. Wave w owns rows 16*(w>>1)..+15 of the 64-row
// tile and half the output columns (w&1). Weight chunks are double-buffered in
// LDS and prefetched by the TDM (wave 0) one chunk ahead of compute.
template <int DI, int DO>
__device__ inline void gemm_tile(const _Float16* __restrict__ sIn,
                                 _Float16* __restrict__ sW,   // [2][HID*64]
                                 const _Float16* __restrict__ wtG,
                                 f8* acc, int tid) {
  constexpr int NC = DI / 64;          // K chunks
  constexpr int CT = DO / 32;          // column tiles per wave
  const int lane = tid & 31;
  const int wave = tid >> 5;
  const int rBase = (wave >> 1) * 16;
  const int colHalf = wave & 1;
  const int aRow = rBase + (lane & 15);
  const int kg   = (lane >> 4) * 8;    // A-frag K-group select
  const int bCol = lane & 15;
  const int kSel = (lane >> 4) * 8;    // B-frag K-group select

  if (wave == 0) tdm_load_chunk<DI, DO>(wtG, sW, 0);

  for (int kc = 0; kc < NC; ++kc) {
    __syncthreads();                   // all waves done with buf[(kc+1)&1]
    if (wave == 0) {
      if (kc + 1 < NC) {
        tdm_load_chunk<DI, DO>(wtG, sW + ((kc + 1) & 1) * (HID * 64), kc + 1);
        __builtin_amdgcn_s_wait_tensorcnt(1);   // chunk kc landed (TDM in-order)
      } else {
        __builtin_amdgcn_s_wait_tensorcnt(0);
      }
    }
    __syncthreads();                   // buf[kc&1] visible to all waves
    const _Float16* wbuf = sW + (kc & 1) * (HID * 64);

#pragma unroll
    for (int ks = 0; ks < 2; ++ks) {
      const int kb = ks * 32;
      // A fragment: 16x32 f16 (lane<16: K {0..7,16..23}; lane>=16: K {8..15,24..31})
      const _Float16* pa = sIn + (size_t)aRow * DI + kc * 64 + kb + kg;
      h8 alo = *(const h8*)pa;
      h8 ahi = *(const h8*)(pa + 16);
      h16 a;
#pragma unroll
      for (int i = 0; i < 8; ++i) { a[i] = alo[i]; a[i + 8] = ahi[i]; }
      // preload all B fragments so ds loads pipeline, then WMMA back-to-back
      h16 bf[CT];
#pragma unroll
      for (int ct = 0; ct < CT; ++ct) {
        const int colBase = (colHalf * CT + ct) * 16;
        const _Float16* pb = wbuf + (size_t)(colBase + bCol) * 64 + kb + kSel;
        h8 blo = *(const h8*)pb;
        h8 bhi = *(const h8*)(pb + 16);
#pragma unroll
        for (int i = 0; i < 8; ++i) { bf[ct][i] = blo[i]; bf[ct][i + 8] = bhi[i]; }
      }
#pragma unroll
      for (int ct = 0; ct < CT; ++ct)
        acc[ct] = __builtin_amdgcn_wmma_f32_16x16x32_f16(
            false, a, false, bf[ct], (short)0, acc[ct], false, false);
    }
  }
  __syncthreads();
}

// bias + ELU epilogue into LDS f16 [MT][DO]
template <int DO>
__device__ inline void epilogue_lds(const f8* acc, const float* __restrict__ bias,
                                    _Float16* __restrict__ sOut, int tid) {
  const int lane = tid & 31;
  const int wave = tid >> 5;
  const int rBase = (wave >> 1) * 16;
  const int colHalf = wave & 1;
  constexpr int CT = DO / 32;
  const int mOff = (lane < 16) ? 0 : 8;   // C layout: VGPR i -> M=i / M=i+8
#pragma unroll
  for (int ct = 0; ct < CT; ++ct) {
    int n = (colHalf * CT + ct) * 16 + (lane & 15);
    float bv = bias[n];
#pragma unroll
    for (int i = 0; i < 8; ++i) {
      int m = rBase + mOff + i;
      float v = acc[ct][i] + bv;
      v = (v > 0.0f) ? v : (__expf(v) - 1.0f);   // ELU
      sOut[m * DO + n] = (_Float16)v;
    }
  }
  __syncthreads();
}

// ---------------------------------------------------------------- fused 3-layer MLP
template <int D_IN, bool GATHER, bool IN_HALF, bool OUT_HALF>
__global__ __launch_bounds__(256, 1) void mlp3_kernel(
    const float* __restrict__ inF, const _Float16* __restrict__ inH, int ldIn,
    const int* __restrict__ gidx,
    const _Float16* __restrict__ wt1, const float* __restrict__ b1,
    const _Float16* __restrict__ wt2, const float* __restrict__ b2,
    const _Float16* __restrict__ wt3, const float* __restrict__ b3,
    _Float16* __restrict__ outH, int ldOut, int colOff,
    float* __restrict__ outF, int nRows) {
  constexpr int SA = (D_IN > HID) ? D_IN : HID;
  __shared__ __align__(16) _Float16 s_a[MT * SA];      // input, later h2
  __shared__ __align__(16) _Float16 s_h1[MT * HID];    // h1
  __shared__ __align__(16) _Float16 s_w[2 * HID * 64]; // double-buffered W chunk

  const int tid = threadIdx.x;
  const int rowBase = blockIdx.x * MT;

  // ---- stage layer-1 input (convert f32->f16, optional gather) ----
  if (IN_HALF) {
    constexpr int U = D_IN / 2;
    for (int i = tid; i < MT * U; i += 256) {
      int r = i / U, c = i - r * U;
      int rr = rowBase + r; if (rr >= nRows) rr = nRows - 1;
      ((uint32_t*)(s_a + r * D_IN))[c] =
          ((const uint32_t*)(inH + (size_t)rr * ldIn))[c];
    }
  } else {
    for (int i = tid; i < MT * D_IN; i += 256) {
      int r = i / D_IN, c = i - r * D_IN;
      int rr = rowBase + r; if (rr >= nRows) rr = nRows - 1;
      int sr = GATHER ? gidx[rr] : rr;
      s_a[r * D_IN + c] = (_Float16)inF[(size_t)sr * D_IN + c];
    }
  }
  // gemm_tile's first __syncthreads() covers the staging above

  f8 acc1[HID / 32] = {};
  gemm_tile<D_IN, HID>(s_a, s_w, wt1, acc1, tid);
  epilogue_lds<HID>(acc1, b1, s_h1, tid);

  f8 acc2[HID / 32] = {};
  gemm_tile<HID, HID>(s_h1, s_w, wt2, acc2, tid);
  epilogue_lds<HID>(acc2, b2, s_a, tid);        // h2 reuses input region

  f8 acc3[OUTD / 32] = {};
  gemm_tile<HID, OUTD>(s_a, s_w, wt3, acc3, tid);

  // ---- final epilogue: bias, no activation, write global ----
  const int lane = tid & 31;
  const int wave = tid >> 5;
  const int rB = (wave >> 1) * 16;
  const int colHalf = wave & 1;
  const int mOff = (lane < 16) ? 0 : 8;
#pragma unroll
  for (int ct = 0; ct < OUTD / 32; ++ct) {
    int n = (colHalf * (OUTD / 32) + ct) * 16 + (lane & 15);
    float bv = b3[n];
#pragma unroll
    for (int i = 0; i < 8; ++i) {
      int row = rowBase + rB + mOff + i;
      if (row < nRows) {
        float v = acc3[ct][i] + bv;
        if (OUT_HALF) outH[(size_t)row * ldOut + colOff + n] = (_Float16)v;
        else          outF[(size_t)row * OUTD + n] = v;
      }
    }
  }
}

// ---------------------------------------------------------------- launch

extern "C" void kernel_launch(void* const* d_in, const int* in_sizes, int n_in,
                              void* d_out, int out_size, void* d_ws, size_t ws_size,
                              hipStream_t stream) {
  (void)n_in; (void)out_size; (void)ws_size;
  const float* items   = (const float*)d_in[0];
  const int*   parents = (const int*)d_in[1];
  const float* ops     = (const float*)d_in[2];
  const int*   iaE     = (const int*)d_in[3];
  const int*   oaE     = (const int*)d_in[4];
  const int N  = in_sizes[0] / 128;
  const int EI = in_sizes[3] / 2;
  const int EO = in_sizes[4] / 2;

  const float* W[5][3]; const float* B[5][3];
  for (int m = 0; m < 5; ++m)
    for (int l = 0; l < 3; ++l) {
      W[m][l] = (const float*)d_in[5 + m * 6 + l * 2];
      B[m][l] = (const float*)d_in[5 + m * 6 + l * 2 + 1];
    }
  const int dins[5] = {128, 128, 128, 64, 512};

  // ---- workspace carving ----
  char* ws = (char*)d_ws;
  size_t off = 0;
  auto carve = [&](size_t bytes) -> char* {
    char* p = ws + off;
    off = (off + bytes + 255) & ~(size_t)255;
    return p;
  };
  float*    aggC = (float*)carve((size_t)N * 128 * sizeof(float));
  float*    aggO = (float*)carve((size_t)N * 64 * sizeof(float));
  _Float16* X    = (_Float16*)carve((size_t)N * 512 * sizeof(_Float16));
  _Float16* wt[5][3];
  for (int m = 0; m < 5; ++m) {
    wt[m][0] = (_Float16*)carve((size_t)dins[m] * HID * 2);
    wt[m][1] = (_Float16*)carve((size_t)HID * HID * 2);
    wt[m][2] = (_Float16*)carve((size_t)HID * OUTD * 2);
  }

  // ---- zero aggregation buffers ----
  zero_f32_kernel<<<2048, 256, 0, stream>>>(aggC, (size_t)N * 128);
  zero_f32_kernel<<<2048, 256, 0, stream>>>(aggO, (size_t)N * 64);

  // ---- weight transpose/convert (f32 [din][dout] -> f16 [dout][din]) ----
  for (int m = 0; m < 5; ++m) {
    int d0 = dins[m];
    transpose_w_kernel<<<(d0 * HID + 255) / 256, 256, 0, stream>>>(W[m][0], wt[m][0], d0, HID);
    transpose_w_kernel<<<(HID * HID + 255) / 256, 256, 0, stream>>>(W[m][1], wt[m][1], HID, HID);
    transpose_w_kernel<<<(HID * OUTD + 255) / 256, 256, 0, stream>>>(W[m][2], wt[m][2], HID, OUTD);
  }

  // ---- segment sums ----
  scatter_add_kernel<128><<<((size_t)EI * 32 + 255) / 256, 256, 0, stream>>>(
      items, iaE, iaE + EI, aggC, EI);
  scatter_add_kernel<64><<<((size_t)EO * 16 + 255) / 256, 256, 0, stream>>>(
      ops, oaE, oaE + EO, aggO, EO);

  // ---- four embedding MLPs -> X[N][512] (parent|children|ops|self) ----
  const int grid = (N + MT - 1) / MT;
  mlp3_kernel<128, false, false, true><<<grid, 256, 0, stream>>>(
      items, nullptr, 0, nullptr,
      wt[0][0], B[0][0], wt[0][1], B[0][1], wt[0][2], B[0][2],
      X, 512, 384, nullptr, N);                                   // self @ 384
  mlp3_kernel<128, true, false, true><<<grid, 256, 0, stream>>>(
      items, nullptr, 0, parents,
      wt[1][0], B[1][0], wt[1][1], B[1][1], wt[1][2], B[1][2],
      X, 512, 0, nullptr, N);                                     // parent @ 0
  mlp3_kernel<128, false, false, true><<<grid, 256, 0, stream>>>(
      aggC, nullptr, 0, nullptr,
      wt[2][0], B[2][0], wt[2][1], B[2][1], wt[2][2], B[2][2],
      X, 512, 128, nullptr, N);                                   // children @ 128
  mlp3_kernel<64, false, false, true><<<grid, 256, 0, stream>>>(
      aggO, nullptr, 0, nullptr,
      wt[3][0], B[3][0], wt[3][1], B[3][1], wt[3][2], B[3][2],
      X, 512, 256, nullptr, N);                                   // ops @ 256

  // ---- combined MLP (f16 in, f32 out) ----
  mlp3_kernel<512, false, true, false><<<grid, 256, 0, stream>>>(
      nullptr, X, 512, nullptr,
      wt[4][0], B[4][0], wt[4][1], B[4][1], wt[4][2], B[4][2],
      nullptr, 0, 0, (float*)d_out, N);

  // ---- reference zeroes the last output row ----
  zero_last_row_kernel<<<1, 128, 0, stream>>>((float*)d_out, N);
}
